// SIMILARITY_79542794322037
// MI455X (gfx1250) — matrix operand order repaired and checked by reading, not compiled, not run on hardware
//
#include <hip/hip_runtime.h>
#include <hip/hip_bf16.h>

#define N_ROWS 131072
#define C_CLS  1000
#define F_DIM  512
#define TILE_C 16
#define NTILES 63          // ceil(1000/16)
#define ROWS_PER_BLOCK 128
#define WT_STRIDE (F_DIM + 16)   // 528 elems -> 1056B rows, 32B aligned

typedef __attribute__((ext_vector_type(16))) __bf16 v16bf;
typedef __attribute__((ext_vector_type(8)))  __bf16 v8bf;
typedef __attribute__((ext_vector_type(8)))  float  v8f;

static __device__ __forceinline__ __bf16 f2bf(float f) {
    union { float f; unsigned u; } v; v.f = f;
    unsigned r = (v.u + 0x7FFFu + ((v.u >> 16) & 1u)) >> 16;   // RNE
    union { unsigned short s; __bf16 b; } o; o.s = (unsigned short)r;
    return o.b;
}

static __device__ __forceinline__ unsigned bfbits(float f) {
    union { float f; unsigned u; } v; v.f = f;
    return ((v.u + 0x7FFFu + ((v.u >> 16) & 1u)) >> 16) & 0xFFFFu;
}

// ---------------------------------------------------------------------------
// Kernel 1: copy init buffers into the output accumulators.
// ---------------------------------------------------------------------------
__global__ void init_kernel(const float* __restrict__ sim_init,
                            const float* __restrict__ cnt_init,
                            float* __restrict__ sim, float* __restrict__ cnt) {
    int i = blockIdx.x * blockDim.x + threadIdx.x;
    if (i < C_CLS * F_DIM) sim[i] = sim_init[i];
    if (i < C_CLS)         cnt[i] = cnt_init[i];
}

// ---------------------------------------------------------------------------
// Kernel 2 (fused): bf16-WMMA logits GEMM + per-row argmax + att_out=0.9*x
//                   + scatter-add of rows/counts into class accumulators.
// 1024 blocks x 256 threads; block owns 128 rows, wave owns 16 rows.
// - LDS W tiles ping-ponged: global loads for t+1 issued before compute of t.
// - B fragments: 2-stage x 2-pair register pipeline feeding two independent
//   WMMA accumulator chains, so ds_load latency overlaps the matrix pipe.
// ---------------------------------------------------------------------------
__global__ __launch_bounds__(256, 1)
void gemm_argmax_scatter_kernel(const float* __restrict__ x,
                                const float* __restrict__ W,
                                const float* __restrict__ bias,
                                float* __restrict__ att,
                                float* __restrict__ sim,
                                float* __restrict__ cnt) {
    __shared__ __bf16 wt[2][TILE_C][WT_STRIDE];   // 2 x 16.5 KB ping-pong
    __shared__ int    bidx[ROWS_PER_BLOCK];       // class id per row

    const int tid  = threadIdx.x;
    const int wave = tid >> 5;
    const int lane = tid & 31;
    const int half = lane >> 4;
    const int ln   = lane & 15;
    const size_t blockRow0 = (size_t)blockIdx.x * ROWS_PER_BLOCK;

    // ---- att_out = 0.9 * x for this block's 128 rows (single HBM pass) ----
    {
        const float4* xin = (const float4*)(x + blockRow0 * F_DIM);
        float4*       ao  = (float4*)(att + blockRow0 * F_DIM);
        #pragma unroll 4
        for (int i = tid; i < ROWS_PER_BLOCK * F_DIM / 4; i += 256) {
            float4 v = xin[i];
            v.x *= 0.9f; v.y *= 0.9f; v.z *= 0.9f; v.w *= 0.9f;
            ao[i] = v;
        }
    }

    // ---- W tile staging helpers (bf16-packed in registers: uint2 x 8) ----
    auto stageLoad = [&](int t, uint2* wv) {
        #pragma unroll
        for (int i = 0; i < 8; ++i) {
            int lin = tid + i * 256;      // float4 index within 16x512 tile
            int r   = lin >> 7;
            int c4  = lin & 127;
            int cls = t * TILE_C + r;
            float4 v = (cls < C_CLS)
                         ? ((const float4*)(W + (size_t)cls * F_DIM))[c4]
                         : make_float4(0.f, 0.f, 0.f, 0.f);
            wv[i].x = bfbits(v.x) | (bfbits(v.y) << 16);
            wv[i].y = bfbits(v.z) | (bfbits(v.w) << 16);
        }
    };
    auto stageStore = [&](int buf, const uint2* wv) {
        #pragma unroll
        for (int i = 0; i < 8; ++i) {
            int lin = tid + i * 256;
            int r   = lin >> 7;
            int c4  = lin & 127;
            *(uint2*)&wt[buf][r][c4 * 4] = wv[i];   // ds_store_b64
        }
    };

    // ---- A fragments: 16 rows x 512 K as bf16, kept in VGPRs -------------
    // A 16x32 layout (matches B): lanes 0-15 hold K 0..15 of row=ln,
    // lanes 16-31 hold K 16..31 of row=ln.
    const int rowbase = (int)blockRow0 + wave * 16;
    const float* xr = x + (size_t)(rowbase + ln) * F_DIM;
    v16bf afr[16];
    #pragma unroll
    for (int kb = 0; kb < 16; ++kb) {
        const int c0 = kb * 32 + half * 16;
        const int c1 = c0 + 8;
        v16bf a;
        #pragma unroll
        for (int j = 0; j < 8; ++j) a[j]     = f2bf(xr[c0 + j]);
        #pragma unroll
        for (int j = 0; j < 8; ++j) a[8 + j] = f2bf(xr[c1 + j]);
        afr[kb] = a;
    }

    float bestv[8];
    int   besti[8];
    #pragma unroll
    for (int e = 0; e < 8; ++e) { bestv[e] = -3.4e38f; besti[e] = 0; }

    // ---- prologue: stage tile 0 ------------------------------------------
    {
        uint2 wv[8];
        stageLoad(0, wv);
        stageStore(0, wv);
    }
    __syncthreads();

    for (int t = 0; t < NTILES; ++t) {
        // Issue global loads for tile t+1 now; they complete under compute.
        uint2 wn[8];
        if (t + 1 < NTILES) {
            __builtin_prefetch(W + (size_t)(t + 1) * TILE_C * F_DIM, 0, 1);
            stageLoad(t + 1, wn);
        }

        // ---- 16x16 logits tile: 16 x v_wmma_f32_16x16x32_bf16 ------------
        // B 32x16 per kb: lanes 0-15 K=kb*32+0..15, lanes 16-31 +16..31,
        // class column = ln. Two-stage register pipeline (4 distinct
        // fragment buffers) + two accumulator chains: WMMA for pair p
        // consumes fragments loaded at pair p-1 while pair p+1's four
        // ds_load_b128 are in flight.
        const __bf16* wrow = &wt[t & 1][ln][half * 16];
        union BU { v16bf v; v8bf h[2]; };
        BU bb[2][2];                       // [stage][even/odd]
        {
            const v8bf* p0 = (const v8bf*)wrow;
            const v8bf* p1 = (const v8bf*)(wrow + 32);
            bb[0][0].h[0] = p0[0]; bb[0][0].h[1] = p0[1];
            bb[0][1].h[0] = p1[0]; bb[0][1].h[1] = p1[1];
        }
        v8f acc0 = {}, acc1 = {};
        #pragma unroll
        for (int p = 0; p < 8; ++p) {      // pair p handles kb = 2p, 2p+1
            const int cur = p & 1;
            const int nxt = cur ^ 1;
            if (p + 1 < 8) {
                const v8bf* p0 = (const v8bf*)(wrow + (2 * p + 2) * 32);
                const v8bf* p1 = (const v8bf*)(wrow + (2 * p + 3) * 32);
                bb[nxt][0].h[0] = p0[0]; bb[nxt][0].h[1] = p0[1];
                bb[nxt][1].h[0] = p1[0]; bb[nxt][1].h[1] = p1[1];
            }
            acc0 = __builtin_amdgcn_wmma_f32_16x16x32_bf16(
                       false, afr[2 * p],     false, bb[cur][0].v,
                       (short)0, acc0, false, false);
            acc1 = __builtin_amdgcn_wmma_f32_16x16x32_bf16(
                       false, afr[2 * p + 1], false, bb[cur][1].v,
                       (short)0, acc1, false, false);
        }

        // C/D layout: lane's column is fixed: col = t*16 + ln; VGPR e is row
        // e (lanes 0-15) or e+8 (lanes 16-31). Running argmax per lane.
        const int col = t * TILE_C + ln;
        if (col < C_CLS) {
            const float bv = bias[col];
            #pragma unroll
            for (int e = 0; e < 8; ++e) {
                float v = acc0[e] + acc1[e] + bv;
                if (v > bestv[e]) { bestv[e] = v; besti[e] = col; }
            }
        }

        __syncthreads();   // all waves done reading wt[(t+1)&1] (iter t-1)
        if (t + 1 < NTILES) stageStore((t + 1) & 1, wn);
        __syncthreads();   // staged tile visible before iteration t+1
    }

    // ---- reduce argmax over the 16 lanes sharing each row ----------------
    #pragma unroll
    for (int e = 0; e < 8; ++e) {
        float bv = bestv[e];
        int   bi = besti[e];
        #pragma unroll
        for (int m = 1; m < 16; m <<= 1) {      // stays within 16-lane halves
            float ov = __shfl_xor(bv, m, 32);
            int   oi = __shfl_xor(bi, m, 32);
            if (ov > bv || (ov == bv && oi < bi)) { bv = ov; bi = oi; }
        }
        if (ln == 0)
            bidx[wave * 16 + e + half * 8] = bi;   // row (rowbase+e+8*half)
    }
    __syncthreads();

    // ---- scatter: similarity[cls] += x[row]; counts[cls] += 1 ------------
    // x strip is L2-hot (just read). 128 rows x 128 float4 per block.
    for (int lin = tid; lin < ROWS_PER_BLOCK * (F_DIM / 4); lin += 256) {
        int r   = lin >> 7;
        int c4  = lin & 127;
        int cls = bidx[r];
        float4 v = ((const float4*)(x + (blockRow0 + r) * F_DIM))[c4];
        float* s = sim + (size_t)cls * F_DIM + c4 * 4;
        atomicAdd(s + 0, v.x);
        atomicAdd(s + 1, v.y);
        atomicAdd(s + 2, v.z);
        atomicAdd(s + 3, v.w);
    }
    if (tid < ROWS_PER_BLOCK)
        atomicAdd(cnt + bidx[tid], 1.0f);
}

// ---------------------------------------------------------------------------
// Kernel 3: similarity_weight = similarity / counts  (inf/nan for empty ok)
// ---------------------------------------------------------------------------
__global__ void weight_kernel(const float* __restrict__ sim,
                              const float* __restrict__ cnt,
                              float* __restrict__ w) {
    int i = blockIdx.x * blockDim.x + threadIdx.x;
    if (i < C_CLS * F_DIM) w[i] = sim[i] / cnt[i / F_DIM];
}

extern "C" void kernel_launch(void* const* d_in, const int* in_sizes, int n_in,
                              void* d_out, int out_size, void* d_ws, size_t ws_size,
                              hipStream_t stream) {
    (void)in_sizes; (void)n_in; (void)out_size; (void)d_ws; (void)ws_size;
    const float* x        = (const float*)d_in[0];
    const float* W        = (const float*)d_in[1];
    const float* b        = (const float*)d_in[2];
    const float* sim_init = (const float*)d_in[3];
    const float* cnt_init = (const float*)d_in[4];

    float* out    = (float*)d_out;
    float* att    = out;                                  // [N,F]
    float* counts = out + (size_t)N_ROWS * F_DIM;         // [C]
    float* sim    = counts + C_CLS;                       // [C,F]
    float* weight = sim + (size_t)C_CLS * F_DIM;          // [C,F]

    {
        int total = C_CLS * F_DIM;
        init_kernel<<<(total + 255) / 256, 256, 0, stream>>>(sim_init, cnt_init,
                                                             sim, counts);
    }
    {
        int blocks = N_ROWS / ROWS_PER_BLOCK;             // 1024
        gemm_argmax_scatter_kernel<<<blocks, 256, 0, stream>>>(x, W, b, att,
                                                               sim, counts);
    }
    {
        int total = C_CLS * F_DIM;
        weight_kernel<<<(total + 255) / 256, 256, 0, stream>>>(sim, counts,
                                                               weight);
    }
}
